// MultiHeadSelfAttention_58506044506913
// MI455X (gfx1250) — compile-verified, matrix operand order checked by reading, not run on hardware
//
#include <hip/hip_runtime.h>
#include <hip/hip_bf16.h>

// ---------------------------------------------------------------------------
// Types for CDNA5 WMMA (wave32): v_wmma_f32_16x16x32_bf16
// ---------------------------------------------------------------------------
typedef __bf16 bf16;
typedef __attribute__((ext_vector_type(8)))  __bf16 v8bf;
typedef __attribute__((ext_vector_type(16))) __bf16 v16bf;
typedef __attribute__((ext_vector_type(8)))  float  v8f;

// Problem constants (match reference)
#define BATCH   4
#define SEQ     2048
#define DMODEL  2048
#define HQ      16
#define HKV     4
#define DHEAD   128
#define QKVN    3072   // (HQ + 2*HKV) * DHEAD

// ---------------------------------------------------------------------------
// LDS staging helpers (convert f32 -> bf16 on the fly, or copy bf16)
// ---------------------------------------------------------------------------
__device__ __forceinline__ void stage8(bf16* dst, const float* src) {
  #pragma unroll
  for (int c = 0; c < 8; c += 4) {
    float4 f = *(const float4*)(src + c);
    dst[c+0] = (bf16)f.x; dst[c+1] = (bf16)f.y;
    dst[c+2] = (bf16)f.z; dst[c+3] = (bf16)f.w;
  }
}
__device__ __forceinline__ void stage8(bf16* dst, const bf16* src) {
  *(v8bf*)(dst) = *(const v8bf*)(src);
}
__device__ __forceinline__ void stage16(bf16* dst, const float* src) {
  stage8(dst, src); stage8(dst + 8, src + 8);
}
__device__ __forceinline__ void stage16(bf16* dst, const bf16* src) {
  stage8(dst, src); stage8(dst + 8, src + 8);
}

// Load a 16-bit WMMA A/B fragment from a row-major (K-contiguous) row.
// ISA layout (16-bit, K=32): lanes 0-15 hold K 0..7 & 16..23,
// lanes 16-31 hold K 8..15 & 24..31.  kh = (lane>=16) ? 8 : 0.
__device__ __forceinline__ v16bf load_frag(const bf16* rowbase, int kh) {
  union { v16bf v; v8bf h[2]; } u;
  u.h[0] = *(const v8bf*)(rowbase + kh);
  u.h[1] = *(const v8bf*)(rowbase + kh + 16);
  return u.v;
}

// 16-lane-group reductions (lanes 0-15 and 16-31 reduce independently,
// matching the C/D layout where lane half selects M rows 0-7 vs 8-15).
__device__ __forceinline__ float rmax16(float v) {
  #pragma unroll
  for (int o = 1; o < 16; o <<= 1) v = fmaxf(v, __shfl_xor(v, o, 32));
  return v;
}
__device__ __forceinline__ float rsum16(float v) {
  #pragma unroll
  for (int o = 1; o < 16; o <<= 1) v += __shfl_xor(v, o, 32);
  return v;
}

// ---------------------------------------------------------------------------
// Generic TN GEMM:  C[M,N] = A[M,K] * B[N,K]^T  (both operands K-contiguous),
// bf16 WMMA with f32 accumulation.  Block tile 256x128, 256 threads (8 waves),
// each wave computes a 64x64 C tile as 4x4 v_wmma_f32_16x16x32_bf16 tiles:
// 16 WMMAs per 32-wide K step against 16 ds_load_b128 fragment loads.
// ---------------------------------------------------------------------------
template <typename TA, typename TB, bool OUT_BF16>
__global__ __launch_bounds__(256) void gemm_tn(const TA* __restrict__ A,
                                               const TB* __restrict__ B,
                                               void* __restrict__ Cp,
                                               int M, int N, int K) {
  // stride 48 elems = 96 B keeps every 16-byte fragment chunk aligned
  __shared__ bf16 As[256][48];
  __shared__ bf16 Bs[128][48];

  const int tid  = threadIdx.x;
  const int lane = tid & 31;
  const int w    = tid >> 5;
  const int lh   = lane & 15;
  const int kh   = (lane >= 16) ? 8 : 0;
  const int wm   = w & 3;       // 4 waves along M (64 rows each)
  const int wn   = w >> 2;      // 2 waves along N (64 cols each)
  const int m0   = blockIdx.y * 256;
  const int n0   = blockIdx.x * 128;

  v8f acc[4][4] = {};

  for (int k0 = 0; k0 < K; k0 += 32) {
    __syncthreads();
    {
      // stage A: 256x32 elems, 32 per thread (one full row)
      const TA* asrc = A + (size_t)(m0 + tid) * K + k0;
      #pragma unroll
      for (int c = 0; c < 32; c += 8) stage8(&As[tid][c], asrc + c);
      // stage B: 128x32 elems, 16 per thread
      int brow = tid >> 1, bcb = (tid & 1) * 16;
      stage16(&Bs[brow][bcb], B + (size_t)(n0 + brow) * K + k0 + bcb);
    }
    __syncthreads();

    v16bf a[4], b[4];
    #pragma unroll
    for (int i = 0; i < 4; i++) a[i] = load_frag(&As[wm * 64 + i * 16 + lh][0], kh);
    #pragma unroll
    for (int j = 0; j < 4; j++) b[j] = load_frag(&Bs[wn * 64 + j * 16 + lh][0], kh);

    #pragma unroll
    for (int i = 0; i < 4; i++)
      #pragma unroll
      for (int j = 0; j < 4; j++)
        acc[i][j] = __builtin_amdgcn_wmma_f32_16x16x32_bf16(
            false, a[i], false, b[j], (short)0, acc[i][j], false, false);
  }

  // C/D layout: lane half selects rows r / r+8; lane%16 selects column.
  #pragma unroll
  for (int i = 0; i < 4; i++)
    #pragma unroll
    for (int j = 0; j < 4; j++) {
      int col = n0 + wn * 64 + j * 16 + lh;
      #pragma unroll
      for (int r = 0; r < 8; r++) {
        int row = m0 + wm * 64 + i * 16 + r + kh;
        if constexpr (OUT_BF16)
          ((bf16*)Cp)[(size_t)row * N + col] = (bf16)acc[i][j][r];
        else
          ((float*)Cp)[(size_t)row * N + col] = acc[i][j][r];
      }
    }
}

// ---------------------------------------------------------------------------
// Flash attention (GQA): one workgroup = (b, h, 64 query rows), 4 waves.
// Each wave owns 16 query rows; online softmax over 64-key tiles.
// 32 WMMAs per key tile (16 QK^T + 16 PV) per wave.
// ---------------------------------------------------------------------------
__global__ __launch_bounds__(128) void attn_kernel(const bf16* __restrict__ qkv,
                                                   const float* __restrict__ mask,
                                                   bf16* __restrict__ out) {
  __shared__ bf16 Ks[64][136];      // K tile: [key][d], 272B row stride (16B mult)
  __shared__ bf16 Vt[128][72];      // V tile transposed: [d][key], 144B stride
  __shared__ bf16 Ps[4][16][64];    // per-wave P tile (C-layout -> A-layout relay)

  const int tid  = threadIdx.x;
  const int lane = tid & 31;
  const int w    = tid >> 5;
  const int lh   = lane & 15;
  const int kh   = (lane >= 16) ? 8 : 0;

  const int nqt = SEQ / 64;
  const int qt  = blockIdx.x % nqt;
  const int h   = (blockIdx.x / nqt) % HQ;
  const int b   = blockIdx.x / (nqt * HQ);
  const int q0  = qt * 64;
  const int hkv = h / (HQ / HKV);

  const bf16* Qb = qkv + (size_t)(b * SEQ) * QKVN + h * DHEAD;
  const bf16* Kb = qkv + (size_t)(b * SEQ) * QKVN + DMODEL + hkv * DHEAD;
  const bf16* Vb = Kb + HKV * DHEAD;   // V block starts DMODEL + HKV*DHEAD in

  // Load this wave's Q fragments once (rows q0 + w*16 + lh, K-dim = d, 4x32)
  v16bf qf[4];
  {
    const bf16* qrow = Qb + (size_t)(q0 + w * 16 + lh) * QKVN;
    #pragma unroll
    for (int kk = 0; kk < 4; kk++) qf[kk] = load_frag(qrow + kk * 32, kh);
  }

  float m[8], l[8];
  v8f o[8] = {};
  #pragma unroll
  for (int r = 0; r < 8; r++) { m[r] = -1e30f; l[r] = 0.0f; }

  const float scale = 0.08838834764831845f;   // 1/sqrt(128)
  const int qrow_base = q0 + w * 16;

  for (int kt = 0; kt < SEQ; kt += 64) {
    __syncthreads();   // previous iteration's LDS reads complete
    {
      int row = tid >> 1, cb = (tid & 1) * 64;   // 64 keys x 128 d
      const bf16* ksrc = Kb + (size_t)(kt + row) * QKVN + cb;
      #pragma unroll
      for (int c = 0; c < 64; c += 8)
        *(v8bf*)(&Ks[row][cb + c]) = *(const v8bf*)(ksrc + c);
      const bf16* vsrc = Vb + (size_t)(kt + row) * QKVN + cb;
      #pragma unroll
      for (int c = 0; c < 64; c += 8) {
        v8bf vv = *(const v8bf*)(vsrc + c);
        #pragma unroll
        for (int e = 0; e < 8; e++) Vt[cb + c + e][row] = vv[e];
      }
    }
    __syncthreads();

    // -------- S = Q * K^T  (four 16x16 key tiles, 4 k-steps over d=128) ----
    v8f s[4] = {};
    #pragma unroll
    for (int kk = 0; kk < 4; kk++) {
      #pragma unroll
      for (int j = 0; j < 4; j++) {
        v16bf bk = load_frag(&Ks[j * 16 + lh][kk * 32], kh);
        s[j] = __builtin_amdgcn_wmma_f32_16x16x32_bf16(false, qf[kk], false, bk,
                                                       (short)0, s[j], false, false);
      }
    }

    // -------- online softmax (per lane: 8 rows = VGPR index + lane-half) ---
    float alpha[8];
    #pragma unroll
    for (int r = 0; r < 8; r++) {
      int qr = qrow_base + r + kh;
      const float* mrow = mask + (size_t)qr * SEQ + kt + lh;
      float v[4], p[4];
      #pragma unroll
      for (int j = 0; j < 4; j++) v[j] = s[j][r] * scale + mrow[j * 16];
      float mx = rmax16(fmaxf(fmaxf(v[0], v[1]), fmaxf(v[2], v[3])));
      float mn = fmaxf(m[r], mx);
      alpha[r] = __expf(m[r] - mn);
      float psum = 0.0f;
      #pragma unroll
      for (int j = 0; j < 4; j++) { p[j] = __expf(v[j] - mn); psum += p[j]; }
      l[r] = l[r] * alpha[r] + rsum16(psum);
      m[r] = mn;
      #pragma unroll
      for (int j = 0; j < 4; j++) Ps[w][r + kh][j * 16 + lh] = (bf16)p[j];
    }
    #pragma unroll
    for (int t = 0; t < 8; t++)
      #pragma unroll
      for (int r = 0; r < 8; r++) o[t][r] *= alpha[r];
    __syncthreads();

    // -------- O += P * V  (K=64 keys = 2 k-steps, 8 d-tiles of 16) ---------
    #pragma unroll
    for (int ks = 0; ks < 2; ks++) {
      v16bf pa = load_frag(&Ps[w][lh][ks * 32], kh);
      #pragma unroll
      for (int t = 0; t < 8; t++) {
        v16bf bv = load_frag(&Vt[t * 16 + lh][ks * 32], kh);
        o[t] = __builtin_amdgcn_wmma_f32_16x16x32_bf16(false, pa, false, bv,
                                                       (short)0, o[t], false, false);
      }
    }
  }

  // -------- epilogue: O /= l, store bf16 into [B*S, DMODEL] ----------------
  bf16* ob = out + (size_t)(b * SEQ + qrow_base) * DMODEL + h * DHEAD;
  #pragma unroll
  for (int r = 0; r < 8; r++) {
    float inv = 1.0f / l[r];
    #pragma unroll
    for (int t = 0; t < 8; t++)
      ob[(size_t)(r + kh) * DMODEL + t * 16 + lh] = (bf16)(o[t][r] * inv);
  }
}

// ---------------------------------------------------------------------------
// Launch: GEMM(qkv proj) -> flash attention -> GEMM(output proj)
// ---------------------------------------------------------------------------
extern "C" void kernel_launch(void* const* d_in, const int* in_sizes, int n_in,
                              void* d_out, int out_size, void* d_ws, size_t ws_size,
                              hipStream_t stream) {
  const float* x    = (const float*)d_in[0];  // [B,S,DMODEL]
  const float* mask = (const float*)d_in[1];  // [S,S]
  const float* Wqkv = (const float*)d_in[2];  // [QKVN, DMODEL]
  const float* Wo   = (const float*)d_in[3];  // [DMODEL, DMODEL]
  float* out        = (float*)d_out;          // [B,S,DMODEL]

  const int M = BATCH * SEQ;                  // 8192
  bf16* qkvb  = (bf16*)d_ws;                  // [M, QKVN]  (48 MB)
  bf16* attnb = qkvb + (size_t)M * QKVN;      // [M, DMODEL] (32 MB)

  // 1) fused QKV projection: qkvb = x @ Wqkv^T  (bf16 out, f32 acc)
  gemm_tn<float, float, true>
      <<<dim3(QKVN / 128, M / 256), 256, 0, stream>>>(x, Wqkv, qkvb, M, QKVN, DMODEL);

  // 2) GQA flash attention: attnb[b,s,h*128+d]
  attn_kernel<<<BATCH * HQ * (SEQ / 64), 128, 0, stream>>>(qkvb, mask, attnb);

  // 3) output projection: out = attnb @ Wo^T  (f32 out)
  gemm_tn<bf16, float, false>
      <<<dim3(DMODEL / 128, M / 256), 256, 0, stream>>>(attnb, Wo, out, M, DMODEL, DMODEL);
}
